// EncoderLayerMOE_36661840839536
// MI455X (gfx1250) — compile-verified, matrix operand order checked by reading, not run on hardware
//
#include <hip/hip_runtime.h>
#include <math.h>

// ---------------------------------------------------------------------------
// Problem constants (from reference)
// ---------------------------------------------------------------------------
enum : int {
  B_ = 2, S_ = 1024, D_ = 512, E_ = 8,
  HMOE_ = 4, HD_ = D_ / HMOE_,       // 128
  HID_ = 4 * D_,                     // 2048
  AH_ = 4, DK_ = D_ / AH_            // 128
};

// ---------------------------------------------------------------------------
// WMMA types (CDNA5 / gfx1250, wave32)
// ---------------------------------------------------------------------------
typedef __bf16 bf16_t;
typedef bf16_t v16bf __attribute__((ext_vector_type(16)));
typedef bf16_t v2bf  __attribute__((ext_vector_type(2)));
typedef float  v8f   __attribute__((ext_vector_type(8)));

static __device__ __forceinline__ unsigned short f2bf(float f) {
  union { bf16_t h; unsigned short u; } c;
  c.h = (bf16_t)f;                    // hardware cvt f32->bf16
  return c.u;
}
static __device__ __forceinline__ unsigned pack2bf(float a, float b) {
#if __has_builtin(__builtin_amdgcn_cvt_pk_bf16_f32)
  union { v2bf v; unsigned u; } c;
  c.v = __builtin_amdgcn_cvt_pk_bf16_f32(a, b);
  return c.u;
#else
  return (unsigned)f2bf(a) | ((unsigned)f2bf(b) << 16);
#endif
}

// ---------------------------------------------------------------------------
// Generic batched/strided bf16 GEMM (all operands pre-converted to bf16,
// B always stored [N x K] so both LDS stages are contiguous uint4 copies):
//   C(fp32) and/or Cb(bf16, optionally transposed) = epi(alpha*A@B^T + bias)
// flags: 1=ReLU  2=accumulate fp32 C  8=write fp32 C  16=write bf16 Cb
//        32=Cb transposed (Cb[n][m], packed 8-wide stores)
// rowscale: value *= rowscale[z*rs_z + m*rs_m]
// ---------------------------------------------------------------------------
#define BM 128
#define BN 128
#define BK 64

struct GemmP {
  const unsigned short* A;   // bf16 [M x K], row stride lda
  const unsigned short* Bm;  // bf16 [N x K], row stride ldb
  const float* bias;         // [N] or null
  float* C;                  // fp32 [M x N], row stride ldc
  unsigned short* Cb;        // bf16 mirror or null
  const float* rowscale;
  int M, N, K, lda, ldb, ldc, ldcb;
  long long sA1, sA2, sB1, sB2, sC1, sC2, sCb1, sCb2;
  int bdiv;
  long long rs_z; int rs_m;
  float alpha;
  int flags;
};

__global__ __launch_bounds__(256)
void gemm_bf16_wmma(GemmP p) {
  __shared__ unsigned short sa[BM * BK];   // [m][k]
  __shared__ unsigned short sb[BN * BK];   // [n][k]

  const int tid  = threadIdx.x;
  const int lane = tid & 31;
  const int wid  = tid >> 5;
  const int wm   = wid >> 2;   // 0..1 -> 64 rows
  const int wn   = wid & 3;    // 0..3 -> 32 cols

  const int m0 = blockIdx.y * BM;
  const int n0 = blockIdx.x * BN;
  const int z  = blockIdx.z;
  const int z1 = z / p.bdiv;
  const int z2 = z % p.bdiv;

  const unsigned short* A  = p.A  + z1 * p.sA1 + z2 * p.sA2;
  const unsigned short* Bm = p.Bm + z1 * p.sB1 + z2 * p.sB2;

  const int l16   = lane & 15;
  const int lhalf = lane >> 4;

  v8f acc[4][2];
  #pragma unroll
  for (int i = 0; i < 4; ++i)
    #pragma unroll
    for (int j = 0; j < 2; ++j)
      #pragma unroll
      for (int r = 0; r < 8; ++r) acc[i][j][r] = 0.f;

  for (int k0 = 0; k0 < p.K; k0 += BK) {
    // ---- stage A tile (128 x 64 bf16): pure uint4 copies ----
    #pragma unroll
    for (int i = 0; i < 4; ++i) {
      int lin = (i * 256 + tid) * 8;        // 0..8191
      int r = lin >> 6, c = lin & 63;       // c % 8 == 0
      uint4 v = make_uint4(0u, 0u, 0u, 0u);
      int gm = m0 + r, gk = k0 + c;
      if (gm < p.M && gk + 7 < p.K) {
        const unsigned short* ap = &A[(long long)gm * p.lda + gk];
        v = *reinterpret_cast<const uint4*>(ap);
        if (i == 0) __builtin_prefetch(ap + 2 * BK, 0, 0);
      }
      *reinterpret_cast<uint4*>(&sa[r * BK + c]) = v;
    }
    // ---- stage B tile (128 x 64 bf16) ----
    #pragma unroll
    for (int i = 0; i < 4; ++i) {
      int lin = (i * 256 + tid) * 8;
      int n = lin >> 6, c = lin & 63;
      uint4 v = make_uint4(0u, 0u, 0u, 0u);
      int gn = n0 + n, gk = k0 + c;
      if (gn < p.N && gk + 7 < p.K) {
        const unsigned short* bp = &Bm[(long long)gn * p.ldb + gk];
        v = *reinterpret_cast<const uint4*>(bp);
        if (i == 0) __builtin_prefetch(bp + 2 * BK, 0, 0);
      }
      *reinterpret_cast<uint4*>(&sb[n * BK + c]) = v;
    }
    __syncthreads();

    // ---- 2 K-steps of 32, 8 WMMA each ----
    union Frag { uint4 q[2]; v16bf v; };
    #pragma unroll
    for (int kk = 0; kk < 2; ++kk) {
      Frag fa[4], fb[2];
      #pragma unroll
      for (int i = 0; i < 4; ++i) {
        int off = (wm * 64 + i * 16 + l16) * BK + kk * 32 + lhalf * 8;
        fa[i].q[0] = *reinterpret_cast<const uint4*>(&sa[off]);
        fa[i].q[1] = *reinterpret_cast<const uint4*>(&sa[off + 16]);
      }
      #pragma unroll
      for (int j = 0; j < 2; ++j) {
        int off = (wn * 32 + j * 16 + l16) * BK + kk * 32 + lhalf * 8;
        fb[j].q[0] = *reinterpret_cast<const uint4*>(&sb[off]);
        fb[j].q[1] = *reinterpret_cast<const uint4*>(&sb[off + 16]);
      }
      #pragma unroll
      for (int i = 0; i < 4; ++i)
        #pragma unroll
        for (int j = 0; j < 2; ++j)
          acc[i][j] = __builtin_amdgcn_wmma_f32_16x16x32_bf16(
              false, fa[i].v, false, fb[j].v, (short)0, acc[i][j],
              false, false);
    }
    __syncthreads();
  }

  // ---- epilogue ----
  const bool relu  = (p.flags & 1)  != 0;
  const bool accum = (p.flags & 2)  != 0;
  const bool wC    = (p.flags & 8)  != 0;
  const bool wCb   = (p.flags & 16) != 0;
  const bool cbT   = (p.flags & 32) != 0;
  const int  mh    = lhalf * 8;

  float* C = p.C + z1 * p.sC1 + z2 * p.sC2;
  unsigned short* Cb = p.Cb ? (p.Cb + z1 * p.sCb1 + z2 * p.sCb2) : nullptr;

  #pragma unroll
  for (int j = 0; j < 2; ++j) {
    int n = n0 + wn * 32 + j * 16 + l16;
    if (n >= p.N) continue;
    float bv = p.bias ? p.bias[n] : 0.f;
    #pragma unroll
    for (int i = 0; i < 4; ++i) {
      int mb = m0 + wm * 64 + i * 16 + mh;   // 8 consecutive rows
      float vv[8];
      #pragma unroll
      for (int r = 0; r < 8; ++r) {
        float v = p.alpha * acc[i][j][r] + bv;
        if (relu) v = fmaxf(v, 0.f);
        int m = mb + r;
        if (p.rowscale && m < p.M)
          v *= p.rowscale[p.rs_z * z + (long long)m * p.rs_m];
        vv[r] = v;
      }
      if (wC) {
        #pragma unroll
        for (int r = 0; r < 8; ++r) {
          int m = mb + r;
          if (m < p.M) {
            float* cp = &C[(long long)m * p.ldc + n];
            *cp = (accum ? *cp : 0.f) + vv[r];
          }
        }
      }
      if (wCb) {
        if (cbT) {
          if (mb + 7 < p.M) {   // packed 8-wide transposed store Cb[n][m..m+7]
            uint4 pk;
            pk.x = pack2bf(vv[0], vv[1]);
            pk.y = pack2bf(vv[2], vv[3]);
            pk.z = pack2bf(vv[4], vv[5]);
            pk.w = pack2bf(vv[6], vv[7]);
            *reinterpret_cast<uint4*>(&Cb[(long long)n * p.ldcb + mb]) = pk;
          } else {
            for (int r = 0; r < 8; ++r)
              if (mb + r < p.M)
                Cb[(long long)n * p.ldcb + mb + r] = f2bf(vv[r]);
          }
        } else {
          #pragma unroll
          for (int r = 0; r < 8; ++r)
            if (mb + r < p.M)
              Cb[(long long)(mb + r) * p.ldcb + n] = f2bf(vv[r]);
        }
      }
    }
  }
}

// ---------------------------------------------------------------------------
// fp32 -> bf16, elementwise (8 per thread)
// ---------------------------------------------------------------------------
__global__ void cvt_bf16_k(const float* in, unsigned short* out,
                           long long n8) {
  long long i = (long long)blockIdx.x * blockDim.x + threadIdx.x;
  if (i >= n8) return;
  const float4* p = reinterpret_cast<const float4*>(in) + i * 2;
  float4 a = p[0], b = p[1];
  uint4 o;
  o.x = pack2bf(a.x, a.y); o.y = pack2bf(a.z, a.w);
  o.z = pack2bf(b.x, b.y); o.w = pack2bf(b.z, b.w);
  reinterpret_cast<uint4*>(out)[i] = o;
}

// ---------------------------------------------------------------------------
// fp32 [R x C] -> bf16 [C x R], batched over blockIdx.z (LDS-tiled 32x32)
// ---------------------------------------------------------------------------
__global__ void cvt_t_bf16_k(const float* in, unsigned short* out,
                             int R, int Ccols) {
  __shared__ unsigned short tile[32][33];
  long long base = (long long)blockIdx.z * R * Ccols;
  int c0 = blockIdx.x * 32, r0 = blockIdx.y * 32;
  for (int rr = threadIdx.y; rr < 32; rr += 8) {
    int r = r0 + rr, c = c0 + threadIdx.x;
    float v = (r < R && c < Ccols) ? in[base + (long long)r * Ccols + c] : 0.f;
    tile[rr][threadIdx.x] = f2bf(v);
  }
  __syncthreads();
  for (int cc = threadIdx.y; cc < 32; cc += 8) {
    int c = c0 + cc, r = r0 + threadIdx.x;
    if (c < Ccols && r < R)
      out[base + (long long)c * R + r] = tile[threadIdx.x][cc];
  }
}

// ---------------------------------------------------------------------------
// Gate + top-2 + re-softmax (E=8)
// ---------------------------------------------------------------------------
__global__ void gate_topk_k(const float* X, const float* GW, const float* GB,
                            float* W, int T, int Sper, int bdiv,
                            long long sx1, long long sx2, int ld, int Kd) {
  int t = blockIdx.x * blockDim.x + threadIdx.x;
  if (t >= T) return;
  int z = t / Sper, s = t % Sper;
  int z1 = z / bdiv, z2 = z % bdiv;
  const float* xr = X + z1 * sx1 + z2 * sx2 + (long long)s * ld;

  float sc[8];
  #pragma unroll
  for (int e = 0; e < 8; ++e) sc[e] = GB[e];
  for (int k2 = 0; k2 < Kd; ++k2) {
    float xv = xr[k2];
    const float* grow = &GW[k2 * 8];
    #pragma unroll
    for (int e = 0; e < 8; ++e) sc[e] += xv * grow[e];
  }
  float mx = sc[0];
  #pragma unroll
  for (int e = 1; e < 8; ++e) mx = fmaxf(mx, sc[e]);
  float pr[8], sum = 0.f;
  #pragma unroll
  for (int e = 0; e < 8; ++e) { pr[e] = __expf(sc[e] - mx); sum += pr[e]; }
  float inv = 1.f / sum;
  #pragma unroll
  for (int e = 0; e < 8; ++e) pr[e] *= inv;

  int i0 = 0; float p0 = pr[0];
  #pragma unroll
  for (int e = 1; e < 8; ++e) if (pr[e] > p0) { p0 = pr[e]; i0 = e; }
  int i1 = (i0 == 0) ? 1 : 0; float p1 = pr[i1];
  #pragma unroll
  for (int e = 0; e < 8; ++e)
    if (e != i0 && pr[e] > p1) { p1 = pr[e]; i1 = e; }

  float m2 = fmaxf(p0, p1);
  float e0 = __expf(p0 - m2), e1 = __expf(p1 - m2);
  float is = 1.f / (e0 + e1);
  float* wr = &W[(long long)t * 8];
  #pragma unroll
  for (int e = 0; e < 8; ++e) wr[e] = 0.f;
  wr[i0] = e0 * is; wr[i1] = e1 * is;
}

// ---------------------------------------------------------------------------
// LayerNorm, one wave32 per row; optional fp32 and bf16 outputs
// ---------------------------------------------------------------------------
__global__ void layernorm_k(const float* X, const float* g, const float* b,
                            float* Yf, unsigned short* Yb, int rows, int D) {
  int row = blockIdx.x * (blockDim.x >> 5) + (threadIdx.x >> 5);
  int lane = threadIdx.x & 31;
  if (row >= rows) return;
  const float* x = X + (long long)row * D;
  float s = 0.f, s2 = 0.f;
  for (int i = lane * 4; i < D; i += 128) {
    float4 v = *reinterpret_cast<const float4*>(&x[i]);
    s  += v.x + v.y + v.z + v.w;
    s2 += v.x * v.x + v.y * v.y + v.z * v.z + v.w * v.w;
  }
  for (int o2 = 16; o2 > 0; o2 >>= 1) {
    s  += __shfl_xor(s, o2, 32);
    s2 += __shfl_xor(s2, o2, 32);
  }
  float mu = s / D;
  float rstd = rsqrtf(s2 / D - mu * mu + 1e-12f);
  for (int i = lane * 4; i < D; i += 128) {
    float4 v  = *reinterpret_cast<const float4*>(&x[i]);
    float4 w4 = *reinterpret_cast<const float4*>(&g[i]);
    float4 b4 = *reinterpret_cast<const float4*>(&b[i]);
    float4 o4;
    o4.x = (v.x - mu) * rstd * w4.x + b4.x;
    o4.y = (v.y - mu) * rstd * w4.y + b4.y;
    o4.z = (v.z - mu) * rstd * w4.z + b4.z;
    o4.w = (v.w - mu) * rstd * w4.w + b4.w;
    if (Yf) *reinterpret_cast<float4*>(&Yf[(long long)row * D + i]) = o4;
    if (Yb) {
      uint2 pk; pk.x = pack2bf(o4.x, o4.y); pk.y = pack2bf(o4.z, o4.w);
      *reinterpret_cast<uint2*>(&Yb[(long long)row * D + i]) = pk;
    }
  }
}

// ---------------------------------------------------------------------------
// Masked softmax, one wave32 per row; fp32 in-place + bf16 mirror
// ---------------------------------------------------------------------------
__global__ void softmax_mask_k(float* Sc, unsigned short* Sb,
                               const unsigned char* mask,
                               int rows, int L, int ahs, int smask) {
  int row = blockIdx.x * (blockDim.x >> 5) + (threadIdx.x >> 5);
  int lane = threadIdx.x & 31;
  if (row >= rows) return;
  float* r = Sc + (long long)row * L;
  const unsigned char* mrow = mask + (long long)(row / ahs) * smask;

  float mx = -3.4e38f;
  for (int i = lane * 4; i < L; i += 128) {
    float4 v = *reinterpret_cast<const float4*>(&r[i]);
    float a0 = mrow[i + 0] ? v.x : -1e9f;
    float a1 = mrow[i + 1] ? v.y : -1e9f;
    float a2 = mrow[i + 2] ? v.z : -1e9f;
    float a3 = mrow[i + 3] ? v.w : -1e9f;
    mx = fmaxf(mx, fmaxf(fmaxf(a0, a1), fmaxf(a2, a3)));
  }
  for (int o2 = 16; o2 > 0; o2 >>= 1) mx = fmaxf(mx, __shfl_xor(mx, o2, 32));

  float sum = 0.f;
  for (int i = lane * 4; i < L; i += 128) {
    float4 v = *reinterpret_cast<const float4*>(&r[i]);
    float4 e4;
    e4.x = mrow[i + 0] ? __expf(v.x - mx) : 0.f;
    e4.y = mrow[i + 1] ? __expf(v.y - mx) : 0.f;
    e4.z = mrow[i + 2] ? __expf(v.z - mx) : 0.f;
    e4.w = mrow[i + 3] ? __expf(v.w - mx) : 0.f;
    sum += e4.x + e4.y + e4.z + e4.w;
    *reinterpret_cast<float4*>(&r[i]) = e4;
  }
  for (int o2 = 16; o2 > 0; o2 >>= 1) sum += __shfl_xor(sum, o2, 32);
  float inv = (sum > 0.f) ? (1.f / sum) : 0.f;
  for (int i = lane * 4; i < L; i += 128) {
    float4 v = *reinterpret_cast<const float4*>(&r[i]);
    v.x *= inv; v.y *= inv; v.z *= inv; v.w *= inv;
    *reinterpret_cast<float4*>(&r[i]) = v;
    uint2 pk; pk.x = pack2bf(v.x, v.y); pk.y = pack2bf(v.z, v.w);
    *reinterpret_cast<uint2*>(&Sb[(long long)row * L + i]) = pk;
  }
}

// ---------------------------------------------------------------------------
// Residual add with dual write
// ---------------------------------------------------------------------------
__global__ void add2_k(const float* a, const float* b, float* o1, float* o2,
                       long long n4) {
  long long i = (long long)blockIdx.x * blockDim.x + threadIdx.x;
  if (i >= n4) return;
  float4 va = reinterpret_cast<const float4*>(a)[i];
  float4 vb = reinterpret_cast<const float4*>(b)[i];
  float4 r = make_float4(va.x + vb.x, va.y + vb.y, va.z + vb.z, va.w + vb.w);
  reinterpret_cast<float4*>(o1)[i] = r;
  if (o2) reinterpret_cast<float4*>(o2)[i] = r;
}

// ---------------------------------------------------------------------------
// Host-side GEMM launcher
// ---------------------------------------------------------------------------
static void gemm(hipStream_t st, const unsigned short* A,
                 const unsigned short* Bm, const float* bias,
                 float* C, unsigned short* Cb, const float* rs,
                 int M, int N, int K, int lda, int ldb, int ldc, int ldcb,
                 int nb, int bdiv,
                 long long sA1, long long sA2, long long sB1, long long sB2,
                 long long sC1, long long sC2, long long sCb1, long long sCb2,
                 long long rs_z, int rs_m, float alpha, int flags) {
  GemmP p;
  p.A = A; p.Bm = Bm; p.bias = bias; p.C = C; p.Cb = Cb; p.rowscale = rs;
  p.M = M; p.N = N; p.K = K;
  p.lda = lda; p.ldb = ldb; p.ldc = ldc; p.ldcb = ldcb;
  p.sA1 = sA1; p.sA2 = sA2; p.sB1 = sB1; p.sB2 = sB2;
  p.sC1 = sC1; p.sC2 = sC2; p.sCb1 = sCb1; p.sCb2 = sCb2;
  p.bdiv = bdiv; p.rs_z = rs_z; p.rs_m = rs_m; p.alpha = alpha; p.flags = flags;
  dim3 g((N + BN - 1) / BN, (M + BM - 1) / BM, nb);
  gemm_bf16_wmma<<<g, dim3(256, 1, 1), 0, st>>>(p);
}

static void cvt(hipStream_t st, const float* in, unsigned short* out,
                size_t n) {
  long long n8 = (long long)(n / 8);
  cvt_bf16_k<<<(unsigned)((n8 + 255) / 256), 256, 0, st>>>(in, out, n8);
}
static void cvtT(hipStream_t st, const float* in, unsigned short* out,
                 int R, int C, int slices) {
  dim3 g((C + 31) / 32, (R + 31) / 32, slices);
  cvt_t_bf16_k<<<g, dim3(32, 8, 1), 0, st>>>(in, out, R, C);
}

// ---------------------------------------------------------------------------
// kernel_launch
// ---------------------------------------------------------------------------
extern "C" void kernel_launch(void* const* d_in, const int* in_sizes, int n_in,
                              void* d_out, int out_size, void* d_ws,
                              size_t ws_size, hipStream_t stream) {
  (void)in_sizes; (void)n_in; (void)out_size; (void)ws_size;

  const float* x        = (const float*)d_in[0];
  const float* pre_w    = (const float*)d_in[1];
  const float* pre_b    = (const float*)d_in[2];
  const float* l_gate_w = (const float*)d_in[3];
  const float* l_gate_b = (const float*)d_in[4];
  const float* l_w1     = (const float*)d_in[5];
  const float* l_b1     = (const float*)d_in[6];
  const float* l_w2     = (const float*)d_in[7];
  const float* l_b2     = (const float*)d_in[8];
  const float* align_w  = (const float*)d_in[9];
  const float* align_b  = (const float*)d_in[10];
  const float* ln1_w    = (const float*)d_in[11];
  const float* ln1_b    = (const float*)d_in[12];
  const float* wq       = (const float*)d_in[13];
  const float* bq       = (const float*)d_in[14];
  const float* wk       = (const float*)d_in[15];
  const float* bk       = (const float*)d_in[16];
  const float* wv       = (const float*)d_in[17];
  const float* bv       = (const float*)d_in[18];
  const float* wo       = (const float*)d_in[19];
  const float* bo       = (const float*)d_in[20];
  const float* ln2_w    = (const float*)d_in[21];
  const float* ln2_b    = (const float*)d_in[22];
  const float* g_gate_w = (const float*)d_in[23];
  const float* g_gate_b = (const float*)d_in[24];
  const float* g_w1     = (const float*)d_in[25];
  const float* g_b1     = (const float*)d_in[26];
  const float* g_w2     = (const float*)d_in[27];
  const float* g_b2     = (const float*)d_in[28];
  const unsigned char* mask = (const unsigned char*)d_in[29];

  float* out = (float*)d_out;

  // ---- workspace bump allocator (256B aligned) ----
  char* wsb = (char*)d_ws;
  size_t off = 0;
  auto allocf = [&](size_t n) {
    float* p = (float*)(wsb + off);
    off = (off + n * 4 + 255) & ~(size_t)255; return p;
  };
  auto alloch = [&](size_t n) {
    unsigned short* p = (unsigned short*)(wsb + off);
    off = (off + n * 2 + 255) & ~(size_t)255; return p;
  };

  const size_t TD  = (size_t)B_ * S_ * D_;            // 1,048,576
  const size_t SS8 = (size_t)B_ * AH_ * S_ * S_;      // 8,388,608
  const size_t HB  = (size_t)B_ * HMOE_ * S_ * HID_;  // 16,777,216

  // fp32 buffers
  float* xp   = allocf(TD);
  float* xacc = allocf(TD);
  float* xo   = allocf(TD);
  float* att  = allocf(TD);
  float* x1   = allocf(TD);
  float* x2   = allocf(TD);
  float* scor = allocf(SS8);
  float* wgtl = allocf((size_t)B_ * HMOE_ * S_ * E_);
  float* wgtg = allocf((size_t)B_ * S_ * E_);
  // bf16 buffers
  unsigned short* xb    = alloch(TD);
  unsigned short* xpb   = alloch(TD);
  unsigned short* xaccb = alloch(TD);
  unsigned short* xnb   = alloch(TD);
  unsigned short* x2b   = alloch(TD);
  unsigned short* qb    = alloch(TD);
  unsigned short* kb    = alloch(TD);
  unsigned short* vTb   = alloch(TD);   // [b][h][d][s]
  unsigned short* ob    = alloch(TD);
  unsigned short* h1b   = alloch(HB);
  unsigned short* attnb = alloch(SS8);
  // bf16 transposed weights [N x K]
  unsigned short* preT   = alloch((size_t)D_ * D_);
  unsigned short* alignT = alloch((size_t)D_ * D_);
  unsigned short* wqT    = alloch((size_t)D_ * D_);
  unsigned short* wkT    = alloch((size_t)D_ * D_);
  unsigned short* wvT    = alloch((size_t)D_ * D_);
  unsigned short* woT    = alloch((size_t)D_ * D_);
  unsigned short* lw1T   = alloch((size_t)E_ * HD_ * HID_);
  unsigned short* lw2T   = alloch((size_t)E_ * HID_ * HD_);
  unsigned short* gw1T   = alloch((size_t)E_ * D_ * HID_);
  unsigned short* gw2T   = alloch((size_t)E_ * HID_ * D_);

  const int Trows = B_ * S_;

  // ---- one-time weight convert+transpose to bf16 [N x K] ----
  cvtT(stream, pre_w,   preT,   D_,   D_,   1);
  cvtT(stream, align_w, alignT, D_,   D_,   1);
  cvtT(stream, wq,      wqT,    D_,   D_,   1);
  cvtT(stream, wk,      wkT,    D_,   D_,   1);
  cvtT(stream, wv,      wvT,    D_,   D_,   1);
  cvtT(stream, wo,      woT,    D_,   D_,   1);
  cvtT(stream, l_w1,    lw1T,   HD_,  HID_, E_);
  cvtT(stream, l_w2,    lw2T,   HID_, HD_,  E_);
  cvtT(stream, g_w1,    gw1T,   D_,   HID_, E_);
  cvtT(stream, g_w2,    gw2T,   HID_, D_,   E_);
  cvt(stream, x, xb, TD);

  // ---- local extractor: xp = x @ pre_w + pre_b (fp32 + bf16 mirror) ----
  gemm(stream, xb, preT, pre_b, xp, xpb, nullptr,
       Trows, D_, D_, D_, D_, D_, D_, 1, 1,
       0, 0, 0, 0, 0, 0, 0, 0, 0, 0, 1.f, 8 | 16);

  // local gate over head-view tokens [B,HMOE,S,HD]
  gate_topk_k<<<(B_ * HMOE_ * S_ + 255) / 256, 256, 0, stream>>>(
      xp, l_gate_w, l_gate_b, wgtl, B_ * HMOE_ * S_, S_, HMOE_,
      (long long)S_ * D_, (long long)HD_, D_, HD_);

  hipMemcpyAsync(xacc, xp, TD * sizeof(float), hipMemcpyDeviceToDevice,
                 stream);

  for (int e = 0; e < E_; ++e) {
    // h1b[z] = relu(xf[z] @ l_w1[e] + l_b1[e]),  z = b*HMOE + h
    gemm(stream, xpb, lw1T + (size_t)e * HD_ * HID_, l_b1 + e * HID_,
         nullptr, h1b, nullptr,
         S_, HID_, HD_, D_, HD_, 0, HID_,
         B_ * HMOE_, HMOE_,
         (long long)S_ * D_, (long long)HD_, 0, 0, 0, 0,
         (long long)HMOE_ * S_ * HID_, (long long)S_ * HID_,
         0, 0, 1.f, 1 | 16);
    // xacc[z] += wgt * (h1b[z] @ l_w2[e] + l_b2[e])
    gemm(stream, h1b, lw2T + (size_t)e * HID_ * HD_, l_b2 + e * HD_,
         xacc, nullptr, wgtl + e,
         S_, HD_, HID_, HID_, HID_, D_, 0,
         B_ * HMOE_, HMOE_,
         (long long)HMOE_ * S_ * HID_, (long long)S_ * HID_, 0, 0,
         (long long)S_ * D_, (long long)HD_, 0, 0,
         (long long)S_ * E_, E_, 1.f, 2 | 8);
  }

  cvt(stream, xacc, xaccb, TD);

  // xo = xacc @ align_w + align_b
  gemm(stream, xaccb, alignT, align_b, xo, nullptr, nullptr,
       Trows, D_, D_, D_, D_, D_, 0, 1, 1,
       0, 0, 0, 0, 0, 0, 0, 0, 0, 0, 1.f, 8);

  // LN1 -> bf16 only
  layernorm_k<<<Trows / 8, 256, 0, stream>>>(xo, ln1_w, ln1_b,
                                             nullptr, xnb, Trows, D_);

  // ---- MHA ----
  gemm(stream, xnb, wqT, bq, nullptr, qb, nullptr,
       Trows, D_, D_, D_, D_, 0, D_, 1, 1,
       0, 0, 0, 0, 0, 0, 0, 0, 0, 0, 1.f, 16);
  gemm(stream, xnb, wkT, bk, nullptr, kb, nullptr,
       Trows, D_, D_, D_, D_, 0, D_, 1, 1,
       0, 0, 0, 0, 0, 0, 0, 0, 0, 0, 1.f, 16);
  // V projection written transposed per head: vTb[b][h][d][s]
  for (int h = 0; h < AH_; ++h) {
    gemm(stream, xnb, wvT + (size_t)h * DK_ * D_, bv + h * DK_,
         nullptr, vTb + (size_t)h * DK_ * S_, nullptr,
         S_, DK_, D_, D_, D_, 0, S_,
         B_, 1,
         (long long)S_ * D_, 0, 0, 0, 0, 0,
         (long long)D_ * S_, 0, 0, 0, 1.f, 16 | 32);
  }

  // scores[b,h] = (Q_bh @ K_bh^T) / sqrt(DK)
  gemm(stream, qb, kb, nullptr, scor, nullptr, nullptr,
       S_, S_, DK_, D_, D_, S_, 0,
       B_ * AH_, AH_,
       (long long)S_ * D_, (long long)DK_,
       (long long)S_ * D_, (long long)DK_,
       (long long)AH_ * S_ * S_, (long long)S_ * S_, 0, 0,
       0, 0, 0.08838834764831845f, 8);

  softmax_mask_k<<<(B_ * AH_ * S_) / 8, 256, 0, stream>>>(
      scor, attnb, mask, B_ * AH_ * S_, S_, AH_ * S_, S_);

  // o[b,h] = attn @ V_bh   (B = vTb, already [d][token])
  gemm(stream, attnb, vTb, nullptr, nullptr, ob, nullptr,
       S_, DK_, S_, S_, S_, 0, D_,
       B_ * AH_, AH_,
       (long long)AH_ * S_ * S_, (long long)S_ * S_,
       (long long)D_ * S_, (long long)DK_ * S_,
       0, 0,
       (long long)S_ * D_, (long long)DK_,
       0, 0, 1.f, 16);

  // att = o @ wo + bo
  gemm(stream, ob, woT, bo, att, nullptr, nullptr,
       Trows, D_, D_, D_, D_, D_, 0, 1, 1,
       0, 0, 0, 0, 0, 0, 0, 0, 0, 0, 1.f, 8);

  // x1 = x + att (also seed d_out with residual for MoE accumulation)
  add2_k<<<(int)(TD / 4 / 256), 256, 0, stream>>>(x, att, x1, out,
                                                  (long long)(TD / 4));

  // LN2 -> fp32 (gate) + bf16 (expert GEMMs)
  layernorm_k<<<Trows / 8, 256, 0, stream>>>(x1, ln2_w, ln2_b,
                                             x2, x2b, Trows, D_);

  gate_topk_k<<<(Trows + 255) / 256, 256, 0, stream>>>(
      x2, g_gate_w, g_gate_b, wgtg, Trows, Trows, 1, 0, 0, D_, D_);

  // ---- global MoE: out += wgt * (relu(x2@W1+b1)@W2 + b2) ----
  for (int e = 0; e < E_; ++e) {
    gemm(stream, x2b, gw1T + (size_t)e * D_ * HID_, g_b1 + e * HID_,
         nullptr, h1b, nullptr,
         Trows, HID_, D_, D_, D_, 0, HID_,
         1, 1, 0, 0, 0, 0, 0, 0, 0, 0, 0, 0, 1.f, 1 | 16);
    gemm(stream, h1b, gw2T + (size_t)e * HID_ * D_, g_b2 + e * D_,
         out, nullptr, wgtg + e,
         Trows, D_, HID_, HID_, HID_, D_, 0,
         1, 1, 0, 0, 0, 0, 0, 0, 0, 0,
         0, E_, 1.f, 2 | 8);
  }
}
